// CustomLayer_45354854646103
// MI455X (gfx1250) — matrix operand order repaired and checked by reading, not compiled, not run on hardware
//
#include <hip/hip_runtime.h>

// ---------------------------------------------------------------------------
// relu(x @ W^T + b) : B=131072, D=256  (all fp32)
// Split-bf16 GEMM on CDNA5 WMMA:  x = x_hi + x_lo, W = W_hi + W_lo (bf16 each)
//   x@W^T ~= x_hi@W_hi + x_hi@W_lo + x_lo@W_hi   (fp32 accum, ~2^-16 rel err)
// W staged in LDS as hi/lo bf16 (264KB of the 320KB WGP LDS, padded rows).
// 512-thread blocks (16 waves) -> 4 waves/SIMD with the whole-WGP LDS footprint.
// Three independent accumulator chains break the WMMA C-RAW serialization.
// ---------------------------------------------------------------------------

typedef __attribute__((ext_vector_type(16))) __bf16 v16bf;
typedef __attribute__((ext_vector_type(8)))  __bf16 v8bf;
typedef __attribute__((ext_vector_type(8)))  float  v8f;

#define DDIM 256
#define LDS_PITCH 264              // 256 + 8 bf16 pad -> conflict-free ds_load_b128
#define NTILES 8192                // 131072 / 16 row-tiles
#define BLOCKS 128
#define THREADS 512
#define WAVES_PER_BLOCK (THREADS / 32)                          // 16
#define TILES_PER_WAVE (NTILES / (BLOCKS * WAVES_PER_BLOCK))    // 4

__global__ __launch_bounds__(THREADS) void neuron_gemm_bf16x3(
    const float* __restrict__ x, const float* __restrict__ W,
    const float* __restrict__ bias, float* __restrict__ out)
{
  extern __shared__ __bf16 smem[];
  __bf16* __restrict__ whi = smem;                     // [256][LDS_PITCH]
  __bf16* __restrict__ wlo = smem + DDIM * LDS_PITCH;  // [256][LDS_PITCH]

  const int tid = threadIdx.x;

  // ---- stage W into LDS as hi/lo bf16 split ------------------------------
  for (int i = tid; i < DDIM * (DDIM / 4); i += THREADS) {
    const int r  = i >> 6;               // 64 float4 per row
    const int c4 = (i & 63) << 2;
    const float4 w4 = *(const float4*)(W + (size_t)r * DDIM + c4);
    const float wv[4] = {w4.x, w4.y, w4.z, w4.w};
#pragma unroll
    for (int j = 0; j < 4; ++j) {
      const __bf16 hh = (__bf16)wv[j];
      whi[r * LDS_PITCH + c4 + j] = hh;
      wlo[r * LDS_PITCH + c4 + j] = (__bf16)(wv[j] - (float)hh);
    }
  }
  __syncthreads();

  const int wave = tid >> 5;
  const int lane = tid & 31;
  const int m    = lane & 15;   // A: row in 16-row tile | B/C: column in 16-col tile
  const int h    = lane >> 4;   // half-wave selector (K / M-half split)

  for (int it = 0; it < TILES_PER_WAVE; ++it) {
    const int t    = (blockIdx.x * WAVES_PER_BLOCK + wave) * TILES_PER_WAVE + it;
    const int row0 = t * 16;
    const float* xrow = x + (size_t)(row0 + m) * DDIM;

    // ---- load 16x256 A tile, split to hi/lo bf16 in WMMA A layout --------
    // 16-bit A 16x32: lanes 0-15 = M0-15; element s holds K = 16h+s (contiguous)
    v16bf ahi[8], alo[8];
#pragma unroll
    for (int kb = 0; kb < 8; ++kb) {
      float arr[16];
      const int ka = kb * 32 + h * 16;
      *(float4*)(arr + 0)  = *(const float4*)(xrow + ka);
      *(float4*)(arr + 4)  = *(const float4*)(xrow + ka + 4);
      *(float4*)(arr + 8)  = *(const float4*)(xrow + ka + 8);
      *(float4*)(arr + 12) = *(const float4*)(xrow + ka + 12);
#pragma unroll
      for (int j = 0; j < 16; ++j) {
        const __bf16 hh = (__bf16)arr[j];
        ahi[kb][j] = hh;
        alo[kb][j] = (__bf16)(arr[j] - (float)hh);
      }
    }

    // ---- prefetch next tile's x rows while we grind WMMAs ----------------
    if (it + 1 < TILES_PER_WAVE) {
      const float* xnext = xrow + (size_t)16 * DDIM;
#pragma unroll
      for (int kb = 0; kb < 8; ++kb)
        __builtin_prefetch(xnext + kb * 32 + h * 8, 0, 0);
    }

    // ---- 16 output tiles of 16 cols each ---------------------------------
#pragma unroll 1
    for (int nt = 0; nt < 16; ++nt) {
      const int N0 = nt * 16;
      v8f c0 = {}, c1 = {}, c2 = {};   // 3 independent accumulation chains
#pragma unroll
      for (int kb = 0; kb < 8; ++kb) {
        // B 32x16 bf16 as used here (contiguous k, as A): lane = 16h + n holds column n, K = 16h + 2v,2v+1
        // -> 16 contiguous bf16 of W row (N0+n) starting at kb*32 + 16h
        const __bf16* bp_h = whi + (N0 + m) * LDS_PITCH + kb * 32 + h * 16;
        const __bf16* bp_l = wlo + (N0 + m) * LDS_PITCH + kb * 32 + h * 16;
        const v8bf bh0 = *(const v8bf*)(bp_h);
        const v8bf bh1 = *(const v8bf*)(bp_h + 8);
        const v8bf bl0 = *(const v8bf*)(bp_l);
        const v8bf bl1 = *(const v8bf*)(bp_l + 8);
        const v16bf bh = __builtin_shufflevector(bh0, bh1,
            0,1,2,3,4,5,6,7,8,9,10,11,12,13,14,15);
        const v16bf bl = __builtin_shufflevector(bl0, bl1,
            0,1,2,3,4,5,6,7,8,9,10,11,12,13,14,15);
        c0 = __builtin_amdgcn_wmma_f32_16x16x32_bf16(false, ahi[kb], false, bh,
                                                     (short)0, c0, false, false);
        c1 = __builtin_amdgcn_wmma_f32_16x16x32_bf16(false, ahi[kb], false, bl,
                                                     (short)0, c1, false, false);
        c2 = __builtin_amdgcn_wmma_f32_16x16x32_bf16(false, alo[kb], false, bh,
                                                     (short)0, c2, false, false);
      }
      // ---- bias + relu + store. C layout: lane = N (mod 16), VGPR r -> M = r + 8h
      const float bn = bias[N0 + m];
      float* op = out + (size_t)row0 * DDIM + N0 + m;
#pragma unroll
      for (int r = 0; r < 8; ++r) {
        float v = (c0[r] + c1[r]) + (c2[r] + bn);
        v = v > 0.0f ? v : 0.0f;
        op[(size_t)(r + 8 * h) * DDIM] = v;
      }
    }
  }
}

extern "C" void kernel_launch(void* const* d_in, const int* in_sizes, int n_in,
                              void* d_out, int out_size, void* d_ws, size_t ws_size,
                              hipStream_t stream) {
  (void)in_sizes; (void)n_in; (void)out_size; (void)d_ws; (void)ws_size;
  const float* x = (const float*)d_in[0];
  const float* W = (const float*)d_in[1];
  const float* b = (const float*)d_in[2];
  float* out = (float*)d_out;

  const size_t shmem = 2ull * DDIM * LDS_PITCH * 2ull;   // 270336 B <= 320KB WGP LDS
  hipFuncSetAttribute((const void*)neuron_gemm_bf16x3,
                      hipFuncAttributeMaxDynamicSharedMemorySize, (int)shmem);
  neuron_gemm_bf16x3<<<dim3(BLOCKS), dim3(THREADS), shmem, stream>>>(x, W, b, out);
}
